// A3C_Model_50706383897350
// MI455X (gfx1250) — compile-verified, matrix operand order checked by reading, not run on hardware
//
#include <hip/hip_runtime.h>
#include <hip/hip_bf16.h>
#include <math.h>

#define N_NODES 100
#define IN_C    65536
#define OUT_C   60
#define ACTION  100
#define N_EDGES 1600

#define MPAD   112            // 7 * 16
#define NPAD   64             // 4 * 16
#define KSTEP  32
#define KCHUNK 1024
#define NCHUNK (IN_C / KCHUNK) // 64
#define GTHREADS 224           // 7 waves: wave == m-tile

typedef __attribute__((ext_vector_type(16))) __bf16 v16bf;
typedef __attribute__((ext_vector_type(8)))  float  v8f;

__device__ __forceinline__ unsigned short f2bf(float f) {
  // round-to-nearest-even f32 -> bf16
  unsigned x = __float_as_uint(f);
  unsigned r = x + 0x7FFFu + ((x >> 16) & 1u);
  return (unsigned short)(r >> 16);
}
__device__ __forceinline__ unsigned pack2bf(float a, float b) {
  return (unsigned)f2bf(a) | ((unsigned)f2bf(b) << 16);
}

// ---------------------------------------------------------------------------
// Kernel 1: degrees -> dinv -> dense normalized adjacency A[dst][src] (MPAD-strided)
// ---------------------------------------------------------------------------
__global__ void build_adj_kernel(const int* __restrict__ ei, float* __restrict__ A) {
  __shared__ float degs[N_NODES];
  __shared__ float dinv[N_NODES];
  const int tid = threadIdx.x;
  const int* src = ei;
  const int* dst = ei + N_EDGES;
  if (tid < N_NODES) degs[tid] = 0.f;
  __syncthreads();
  for (int e = tid; e < N_EDGES; e += blockDim.x)
    atomicAdd(&degs[src[e]], 1.0f);
  __syncthreads();
  if (tid < N_NODES) {
    float d = degs[tid];
    dinv[tid] = (d > 0.f) ? rsqrtf(fmaxf(d, 1.0f)) : 0.f;
  }
  for (int i = tid; i < MPAD * MPAD; i += blockDim.x) A[i] = 0.f;
  __syncthreads();
  for (int e = tid; e < N_EDGES; e += blockDim.x) {
    int s = src[e], d = dst[e];
    atomicAdd(&A[d * MPAD + s], -dinv[s] * dinv[d]);
  }
}

// ---------------------------------------------------------------------------
// Kernel 2/3: OUT[n,c] = alpha * sum_m A[n,m] * X[m,c] + beta * SUB[n,c]
// One thread per column c; A cached in LDS; x rows L2-resident.
// ---------------------------------------------------------------------------
__global__ void __launch_bounds__(256)
propagate_kernel(const float* __restrict__ A, const float* __restrict__ X,
                 const float* __restrict__ SUB, float* __restrict__ OUT,
                 float alpha, float beta) {
  __shared__ float As[N_NODES * N_NODES]; // 40 KB
  const int tid = threadIdx.x;
  for (int i = tid; i < N_NODES * N_NODES; i += blockDim.x) {
    int n = i / N_NODES, m = i % N_NODES;
    As[i] = A[n * MPAD + m];
  }
  __syncthreads();
  const int c = blockIdx.x * blockDim.x + tid;
  for (int n0 = 0; n0 < N_NODES; n0 += 10) {
    float acc[10];
    #pragma unroll
    for (int i = 0; i < 10; ++i) acc[i] = 0.f;
    for (int m = 0; m < N_NODES; ++m) {
      float xv = X[(size_t)m * IN_C + c];
      #pragma unroll
      for (int i = 0; i < 10; ++i)
        acc[i] = fmaf(As[(n0 + i) * N_NODES + m], xv, acc[i]);
    }
    #pragma unroll
    for (int i = 0; i < 10; ++i) {
      float r = alpha * acc[i];
      if (SUB) r = fmaf(beta, SUB[(size_t)(n0 + i) * IN_C + c], r);
      OUT[(size_t)(n0 + i) * IN_C + c] = r;
    }
  }
}

// ---------------------------------------------------------------------------
// Kernel 4: split-K WMMA GEMM. block = (head h, slice s, k-chunk).
// 7 waves, wave == m-tile; each wave owns all 4 n-tiles (branch-free, EXEC
// stays all-ones around every WMMA; A-fragment reused across n-tiles).
// Y[h] += Z_s[0:112, kchunk] @ W[h][s][kchunk, 0:64]   (bf16 in, f32 accum)
// ---------------------------------------------------------------------------
__global__ void __launch_bounds__(GTHREADS)
cheb_gemm_wmma_kernel(const float* __restrict__ X,  const float* __restrict__ Z1,
                      const float* __restrict__ Z2, const float* __restrict__ Wa,
                      const float* __restrict__ Wc, float* __restrict__ Y) {
  __shared__ unsigned short As[MPAD * KSTEP]; // [m][k] bf16
  __shared__ unsigned short Bs[NPAD * KSTEP]; // [n][k] bf16 (transposed W tile)

  const int tid  = threadIdx.x;
  const int wave = tid >> 5;      // == m-tile, 0..6
  const int lane = tid & 31;
  const int half = lane >> 4;
  const int ml   = lane & 15;

  const int chunk = blockIdx.x % NCHUNK;
  const int pair  = blockIdx.x / NCHUNK;
  const int h = pair / 3;
  const int s = pair % 3;
  const float* Z = (s == 0) ? X : (s == 1) ? Z1 : Z2;
  const float* W = ((h == 0) ? Wa : Wc) + (size_t)s * IN_C * OUT_C;
  const int kbase = chunk * KCHUNK;

  v8f accs[4];
  #pragma unroll
  for (int t = 0; t < 4; ++t)
    #pragma unroll
    for (int i = 0; i < 8; ++i) accs[t][i] = 0.f;

  const unsigned* arow = (const unsigned*)(As + (wave * 16 + ml) * KSTEP);
  const int base = half * 4;       // dword d holds K = 2d; halves per ISA layout

  for (int k0 = kbase; k0 < kbase + KCHUNK; k0 += KSTEP) {
    // stage A panel as float4 quads: 112*32/4 = 896 quads = 4 per thread exact.
    // quad v: m = v>>3, k = (v&7)*4; 16B-aligned global loads, 8B LDS stores.
    #pragma unroll
    for (int i = 0; i < (MPAD * KSTEP / 4) / GTHREADS; ++i) {
      int v = tid + GTHREADS * i;
      int m = v >> 3, k = (v & 7) * 4;
      float4 q = make_float4(0.f, 0.f, 0.f, 0.f);
      if (m < N_NODES)
        q = *(const float4*)(Z + (size_t)m * IN_C + (k0 + k));
      uint2 p;
      p.x = pack2bf(q.x, q.y);
      p.y = pack2bf(q.z, q.w);
      *(uint2*)(As + m * KSTEP + k) = p;
    }
    // stage B panel transposed: quad task = 4 consecutive k for one column n.
    // 64 n * 8 kq = 512 quads; 4 coalesced b32 loads -> one 8B LDS store.
    for (int v = tid; v < (NPAD * KSTEP) / 4; v += GTHREADS) {
      int n = v & 63, kq = (v >> 6) * 4;
      float q0 = 0.f, q1 = 0.f, q2 = 0.f, q3 = 0.f;
      if (n < OUT_C) {
        const float* wp = W + (size_t)(k0 + kq) * OUT_C + n;
        q0 = wp[0]; q1 = wp[OUT_C]; q2 = wp[2 * OUT_C]; q3 = wp[3 * OUT_C];
      }
      uint2 p;
      p.x = pack2bf(q0, q1);
      p.y = pack2bf(q2, q3);
      *(uint2*)(Bs + n * KSTEP + kq) = p;
    }
    if (k0 + KSTEP < kbase + KCHUNK)
      __builtin_prefetch(&W[(size_t)(k0 + KSTEP + (tid >> 3)) * OUT_C], 0, 1);
    __syncthreads();

    // A fragment once per wave, reused across all 4 n-tiles
    union { v16bf v; unsigned u[8]; } af;
    #pragma unroll
    for (int j = 0; j < 4; ++j) {
      af.u[j]     = arow[base + j];      // K = 8*half + 2j
      af.u[j + 4] = arow[8 + base + j];  // K = 16 + 8*half + 2j
    }
    #pragma unroll
    for (int nt = 0; nt < 4; ++nt) {
      union { v16bf v; unsigned u[8]; } bf;
      const unsigned* brow = (const unsigned*)(Bs + (nt * 16 + ml) * KSTEP);
      #pragma unroll
      for (int j = 0; j < 4; ++j) {
        bf.u[j]     = brow[base + j];
        bf.u[j + 4] = brow[8 + base + j];
      }
      accs[nt] = __builtin_amdgcn_wmma_f32_16x16x32_bf16(
          false, af.v, false, bf.v, (short)0, accs[nt], false, false);
    }
    __syncthreads();
  }

  // scatter-accumulate partials (f32 global atomics into tiny Y buffer)
  float* Yh = Y + (size_t)h * MPAD * NPAD;
  #pragma unroll
  for (int nt = 0; nt < 4; ++nt) {
    #pragma unroll
    for (int rr = 0; rr < 8; ++rr) {
      int m = wave * 16 + rr + half * 8; // C layout: VGPR rr -> M=rr (+8 upper half)
      int n = nt * 16 + ml;
      atomicAdd(&Yh[m * NPAD + n], accs[nt][rr]);
    }
  }
}

// ---------------------------------------------------------------------------
// Kernel 5: tanh + bias + vnr broadcast, then the two tiny FC heads.
// ---------------------------------------------------------------------------
__global__ void __launch_bounds__(256)
head_kernel(const float* __restrict__ Y, const float* __restrict__ vnr,
            const float* __restrict__ acb, const float* __restrict__ ccb,
            const float* __restrict__ avw, const float* __restrict__ avb,
            const float* __restrict__ cvw, const float* __restrict__ cvb,
            const float* __restrict__ afw, const float* __restrict__ afb,
            const float* __restrict__ cfw, const float* __restrict__ cfb,
            float* __restrict__ out) {
  __shared__ float embA[N_NODES * OUT_C];
  __shared__ float embC[N_NODES * OUT_C];
  __shared__ float vsA[OUT_C], vsC[OUT_C];
  __shared__ float red[256];
  const int tid = threadIdx.x;
  const float v0 = vnr[0], v1 = vnr[1], v2 = vnr[2];
  if (tid < OUT_C) {
    vsA[tid] = v0 * avw[tid]       + avb[tid]
             + v1 * avw[60 + tid]  + avb[60 + tid]
             + v2 * avw[120 + tid] + avb[120 + tid];
    vsC[tid] = v0 * cvw[tid]       + cvb[tid]
             + v1 * cvw[60 + tid]  + cvb[60 + tid]
             + v2 * cvw[120 + tid] + cvb[120 + tid];
  }
  __syncthreads();
  for (int i = tid; i < N_NODES * OUT_C; i += 256) {
    int m = i / OUT_C, n = i % OUT_C;
    embA[i] = tanhf(Y[m * NPAD + n] + acb[n]) + vsA[n];
    embC[i] = tanhf(Y[MPAD * NPAD + m * NPAD + n] + ccb[n]) + vsC[n];
  }
  __syncthreads();
  if (tid < ACTION) {
    float acc = afb[tid];
    for (int i = 0; i < N_NODES * OUT_C; ++i)
      acc = fmaf(embA[i], afw[i * ACTION + tid], acc);
    out[tid] = acc;
  }
  float p = 0.f;
  for (int i = tid; i < N_NODES * OUT_C; i += 256)
    p = fmaf(embC[i], cfw[i], p);
  red[tid] = p;
  __syncthreads();
  for (int off = 128; off > 0; off >>= 1) {
    if (tid < off) red[tid] += red[tid + off];
    __syncthreads();
  }
  if (tid == 0) out[ACTION] = red[0] + cfb[0];
}

// ---------------------------------------------------------------------------
extern "C" void kernel_launch(void* const* d_in, const int* in_sizes, int n_in,
                              void* d_out, int out_size, void* d_ws, size_t ws_size,
                              hipStream_t stream) {
  (void)in_sizes; (void)n_in; (void)out_size; (void)ws_size;
  const float* x   = (const float*)d_in[0];
  const int*   ei  = (const int*)d_in[1];
  const float* vnr = (const float*)d_in[2];
  const float* acw = (const float*)d_in[3];
  const float* acb = (const float*)d_in[4];
  const float* ccw = (const float*)d_in[5];
  const float* ccb = (const float*)d_in[6];
  const float* avw = (const float*)d_in[7];
  const float* avb = (const float*)d_in[8];
  const float* cvw = (const float*)d_in[9];
  const float* cvb = (const float*)d_in[10];
  const float* afw = (const float*)d_in[11];
  const float* afb = (const float*)d_in[12];
  const float* cfw = (const float*)d_in[13];
  const float* cfb = (const float*)d_in[14];
  float* out = (float*)d_out;

  // workspace layout (floats): z1 | z2 | A[112*112] | Y[2*112*64]
  float* z1 = (float*)d_ws;
  float* z2 = z1 + (size_t)N_NODES * IN_C;
  float* A  = z2 + (size_t)N_NODES * IN_C;
  float* Y  = A + MPAD * MPAD;

  hipMemsetAsync(Y, 0, 2 * MPAD * NPAD * sizeof(float), stream);
  build_adj_kernel<<<1, 256, 0, stream>>>(ei, A);
  propagate_kernel<<<IN_C / 256, 256, 0, stream>>>(A, x, nullptr, z1, 1.f, 0.f);
  propagate_kernel<<<IN_C / 256, 256, 0, stream>>>(A, z1, x, z2, 2.f, -1.f);
  cheb_gemm_wmma_kernel<<<6 * NCHUNK, GTHREADS, 0, stream>>>(x, z1, z2, acw, ccw, Y);
  head_kernel<<<1, 256, 0, stream>>>(Y, vnr, acb, ccb, avw, avb, cvw, cvb,
                                     afw, afb, cfw, cfb, out);
}
